// Attention_87565793230859
// MI455X (gfx1250) — matrix operand order repaired and checked by reading, not compiled, not run on hardware
//
#include <hip/hip_runtime.h>
#include <hip/hip_bf16.h>

// ---------------------------------------------------------------------------
// Attention (B=8, N=2048, DIM=768, H=12, Dh=64) for MI455X / gfx1250.
// All matmuls on v_wmma_f32_16x16x32_bf16 (fp32 accumulate).
// All LDS fragment loads are contiguous-per-lane (2x ds_load_b128 each);
// staging uses GLOBAL_LOAD_ASYNC_TO_LDS_B128 when the builtin exists.
// ---------------------------------------------------------------------------

typedef __attribute__((ext_vector_type(16))) __bf16 v16bf;
typedef __attribute__((ext_vector_type(8)))  float  v8f;
typedef int v4i_vs __attribute__((vector_size(16)));   // builtin param type

#define BATCH     8
#define SEQ       2048
#define DIM       768
#define NUM_HEADS 12
#define HEAD_DIM  64
#define SCALE_Q   0.125f   // 64^-0.5

// ----------------------- async copy to LDS (guarded) ------------------------

#if defined(__HIP_DEVICE_COMPILE__) && \
    __has_builtin(__builtin_amdgcn_global_load_async_to_lds_b128)
#define USE_ASYNC_LDS 1
#endif

// copy 16 bytes global -> LDS (per active lane)
static __device__ __forceinline__ void copy16_to_lds(const void* g, void* l) {
#ifdef USE_ASYNC_LDS
  __builtin_amdgcn_global_load_async_to_lds_b128((v4i_vs*)g, (v4i_vs*)l, 0, 0);
#else
  *(uint4*)l = *(const uint4*)g;
#endif
}

static __device__ __forceinline__ void copy_wait() {
#ifdef USE_ASYNC_LDS
#if __has_builtin(__builtin_amdgcn_s_wait_asynccnt)
  __builtin_amdgcn_s_wait_asynccnt(0);
#else
  asm volatile("s_wait_asynccnt 0" ::: "memory");
#endif
#endif
}

// ---------------- WMMA fragment loaders (contiguous-k order) ----------------

// A fragment, 16x32 bf16, source row-major (m,k) = p[m*ld + k].
// lanes 0-15: rows 0-15, K = 0..15; lanes 16-31: K = 16..31.
// Per lane: one contiguous 32B run -> 2x ds_load_b128.
static __device__ __forceinline__ v16bf frag_a(const __bf16* p, int ld) {
  int lane = threadIdx.x & 31;
  int m    = lane & 15;
  int kb   = (lane >> 4) << 4;          // 0 or 16
  v16bf a;
#pragma unroll
  for (int i = 0; i < 16; ++i) {
    int k = kb + i;
    a[i] = p[m * ld + k];
  }
  return a;
}

// B fragment, 32x16 bf16, where source is stored N-major: (k,n) = p[n*ld + k].
// lanes 0-15: cols, K = 0..15; lanes 16-31: cols, K = 16..31.
// Per lane: one contiguous 32B run -> 2x ds_load_b128.
static __device__ __forceinline__ v16bf frag_bT(const __bf16* p, int ld) {
  int lane = threadIdx.x & 31;
  int n    = lane & 15;
  int kb   = (lane >> 4) << 4;          // 0 or 16
  v16bf b;
#pragma unroll
  for (int i = 0; i < 16; ++i) b[i] = p[n * ld + kb + i];
  return b;
}

static __device__ __forceinline__ v8f wmma_bf16(v16bf a, v16bf b, v8f c) {
  return __builtin_amdgcn_wmma_f32_16x16x32_bf16(false, a, false, b,
                                                 (short)0, c, false, false);
}

// ---------------------------- fp32 -> bf16 cast -----------------------------

__global__ void cvt_f32_bf16_kernel(const float* __restrict__ in,
                                    __bf16* __restrict__ out, int n) {
  int i = blockIdx.x * blockDim.x + threadIdx.x;
  if (i < n) out[i] = (__bf16)in[i];
}

// ------------------------------- WMMA GEMM ---------------------------------
// C[M,N] = A[M,K] @ B[K,N]; workgroup tile 128x128, 8 waves (4x2), wave tile
// 32x64 = 2x4 WMMA tiles; K staged 32-deep. A staged row-major (async copy),
// B staged transposed [n][k] so fragments load contiguously.
// mode 0: scatter into q/k/v per-head layout (scale on q); mode 1: f32 + bias.
__global__ __launch_bounds__(256) void gemm_bf16_kernel(
    const __bf16* __restrict__ A, const __bf16* __restrict__ B,
    int K, int N, int nTilesN, int mode,
    __bf16* __restrict__ qw, __bf16* __restrict__ kw, __bf16* __restrict__ vw,
    const float* __restrict__ bias, float* __restrict__ out) {
  __shared__ __bf16 As[128 * 32];    // [m][k]
  __shared__ __bf16 BsT[128 * 32];   // [n][k]

  int tid   = threadIdx.x;
  int lane  = tid & 31;
  int wave  = tid >> 5;
  int waveM = wave >> 1;   // 0..3
  int waveN = wave & 1;    // 0..1
  int tileN = blockIdx.x % nTilesN;
  int tileM = blockIdx.x / nTilesN;
  int m0 = tileM * 128, n0 = tileN * 128;

  v8f zero = {};
  v8f acc[2][4];
#pragma unroll
  for (int ms = 0; ms < 2; ++ms)
#pragma unroll
    for (int ns = 0; ns < 4; ++ns) acc[ms][ns] = zero;

  for (int k0 = 0; k0 < K; k0 += 32) {
    // A tile 128x32: 512 16B-chunks, direct async copy (no transform)
#pragma unroll
    for (int i = 0; i < 2; ++i) {
      int idx = tid + i * 256;
      int ar = idx >> 2, aq = idx & 3;
      copy16_to_lds(A + (size_t)(m0 + ar) * K + k0 + aq * 8,
                    As + ar * 32 + aq * 8);
    }
    // B tile 32x128 -> BsT[n][k]: transpose scatter
#pragma unroll
    for (int i = 0; i < 2; ++i) {
      int idx = tid + i * 256;
      int kr = idx >> 4, nq = idx & 15;
      uint4 t = *(const uint4*)(B + (size_t)(k0 + kr) * N + n0 + nq * 8);
      const __bf16* e = (const __bf16*)&t;
#pragma unroll
      for (int j = 0; j < 8; ++j) BsT[(nq * 8 + j) * 32 + kr] = e[j];
    }
    copy_wait();
    __syncthreads();

    v16bf am[2], bn[4];
#pragma unroll
    for (int ms = 0; ms < 2; ++ms)
      am[ms] = frag_a(As + (waveM * 32 + ms * 16) * 32, 32);
#pragma unroll
    for (int ns = 0; ns < 4; ++ns)
      bn[ns] = frag_bT(BsT + (waveN * 64 + ns * 16) * 32, 32);
#pragma unroll
    for (int ms = 0; ms < 2; ++ms)
#pragma unroll
      for (int ns = 0; ns < 4; ++ns)
        acc[ms][ns] = wmma_bf16(am[ms], bn[ns], acc[ms][ns]);
    __syncthreads();
  }

  // epilogue: D element r -> (row = r + 8*(lane>=16), col = lane&15)
  int hi = lane >> 4, col = lane & 15;
#pragma unroll
  for (int ms = 0; ms < 2; ++ms) {
#pragma unroll
    for (int ns = 0; ns < 4; ++ns) {
#pragma unroll
      for (int r = 0; r < 8; ++r) {
        int grow = m0 + waveM * 32 + ms * 16 + r + 8 * hi;
        int gcol = n0 + waveN * 64 + ns * 16 + col;
        float val = acc[ms][ns][r];
        if (mode == 0) {
          int three = gcol / DIM;
          int rem   = gcol - three * DIM;
          int h = rem >> 6, d = rem & 63;
          int bb = grow >> 11, nn = grow & (SEQ - 1);
          size_t off =
              (((size_t)bb * NUM_HEADS + h) * SEQ + nn) * HEAD_DIM + d;
          if (three == 0)      qw[off] = (__bf16)(val * SCALE_Q);
          else if (three == 1) kw[off] = (__bf16)val;
          else                 vw[off] = (__bf16)val;
        } else {
          out[(size_t)grow * DIM + gcol] = val + bias[gcol];
        }
      }
    }
  }
}

// ---------------------------- Flash attention ------------------------------
// One wave32 per (b, h, 16-query block). Key blocks of 32; online softmax.
// K tile staged row-major (async, fragments read contiguous d-runs);
// V tile staged transposed [d][key] so PV fragments read contiguous key-runs.
__global__ __launch_bounds__(32) void flash_attn_kernel(
    const __bf16* __restrict__ q, const __bf16* __restrict__ k,
    const __bf16* __restrict__ v, const int* __restrict__ mask,
    __bf16* __restrict__ attn_out) {
  __shared__ __bf16 kt[32 * HEAD_DIM];   // [key][d]
  __shared__ __bf16 vtT[HEAD_DIM * 32];  // [d][key]
  __shared__ __bf16 pt[16 * 32];         // P transpose staging

  int lane = threadIdx.x & 31;
  int wg = blockIdx.x;
  int qb = wg & (SEQ / 16 - 1);               // 0..127
  int h  = (wg >> 7) % NUM_HEADS;
  int b  = wg / ((SEQ / 16) * NUM_HEADS);

  size_t headBase = ((size_t)b * NUM_HEADS + h) * SEQ * HEAD_DIM;
  const __bf16* qp = q + headBase + (size_t)qb * 16 * HEAD_DIM;

  v16bf qf0 = frag_a(qp, HEAD_DIM);        // d 0..31  (scale pre-folded)
  v16bf qf1 = frag_a(qp + 32, HEAD_DIM);   // d 32..63

  v8f zero = {};
  v8f o0 = zero, o1 = zero, o2 = zero, o3 = zero;
  float mrow[8], lrow[8];
#pragma unroll
  for (int r = 0; r < 8; ++r) { mrow[r] = -1e30f; lrow[r] = 0.f; }

  const int* maskRow = mask + (size_t)b * SEQ;

  for (int kb = 0; kb < SEQ / 32; ++kb) {
    const __bf16* kg = k + headBase + (size_t)kb * 32 * HEAD_DIM;
    const __bf16* vg = v + headBase + (size_t)kb * 32 * HEAD_DIM;
    // K tile: 256 16B chunks, direct async copy
#pragma unroll
    for (int i = 0; i < 8; ++i) {
      int u = lane + 32 * i;
      copy16_to_lds(kg + u * 8, kt + u * 8);
    }
    // V tile -> vtT[d][key]: transpose scatter
#pragma unroll
    for (int i = 0; i < 8; ++i) {
      int u = lane + 32 * i;
      int key = u >> 3, dg = u & 7;
      uint4 t = *(const uint4*)(vg + u * 8);
      const __bf16* e = (const __bf16*)&t;
#pragma unroll
      for (int j = 0; j < 8; ++j) vtT[(dg * 8 + j) * 32 + key] = e[j];
    }
    copy_wait();
    __syncthreads();

    // scores: two 16x16 tiles (keys 0..15 / 16..31), contract d=64 in 2 chunks
    v8f s0 = zero, s1 = zero;
    s0 = wmma_bf16(qf0, frag_bT(kt, HEAD_DIM), s0);
    s0 = wmma_bf16(qf1, frag_bT(kt + 32, HEAD_DIM), s0);
    s1 = wmma_bf16(qf0, frag_bT(kt + 16 * HEAD_DIM, HEAD_DIM), s1);
    s1 = wmma_bf16(qf1, frag_bT(kt + 16 * HEAD_DIM + 32, HEAD_DIM), s1);

    // key mask (column == key within tile)
    int key0 = kb * 32 + (lane & 15);
    bool ok0 = maskRow[key0] != 0;
    bool ok1 = maskRow[key0 + 16] != 0;
#pragma unroll
    for (int r = 0; r < 8; ++r) {
      if (!ok0) s0[r] = -1e30f;
      if (!ok1) s1[r] = -1e30f;
    }

    // online softmax; D-tile rows live across 16-lane half-wave groups
    int hi = lane >> 4, col = lane & 15;
#pragma unroll
    for (int r = 0; r < 8; ++r) {
      float rm = fmaxf(s0[r], s1[r]);
#pragma unroll
      for (int off = 8; off >= 1; off >>= 1)
        rm = fmaxf(rm, __shfl_xor(rm, off));
      float mn    = fmaxf(mrow[r], rm);
      float alpha = __expf(mrow[r] - mn);
      mrow[r] = mn;
      float p0 = __expf(s0[r] - mn);
      float p1 = __expf(s1[r] - mn);
      float rs = p0 + p1;
#pragma unroll
      for (int off = 8; off >= 1; off >>= 1) rs += __shfl_xor(rs, off);
      lrow[r] = lrow[r] * alpha + rs;
      o0[r] *= alpha; o1[r] *= alpha; o2[r] *= alpha; o3[r] *= alpha;
      s0[r] = p0; s1[r] = p1;
    }

    // transpose P (D layout -> A layout) through LDS
#pragma unroll
    for (int r = 0; r < 8; ++r) {
      pt[(r + 8 * hi) * 32 + col]      = (__bf16)s0[r];
      pt[(r + 8 * hi) * 32 + 16 + col] = (__bf16)s1[r];
    }
    __syncthreads();
    v16bf pa = frag_a(pt, 32);

    // O += P @ V  (contract 32 keys; 4 d-chunks of 16)
    o0 = wmma_bf16(pa, frag_bT(vtT + 0 * 32, 32), o0);
    o1 = wmma_bf16(pa, frag_bT(vtT + 16 * 32, 32), o1);
    o2 = wmma_bf16(pa, frag_bT(vtT + 32 * 32, 32), o2);
    o3 = wmma_bf16(pa, frag_bT(vtT + 48 * 32, 32), o3);
    __syncthreads();
  }

  // normalize and store to [B, N, H*64] bf16 for the projection GEMM
  int hi = lane >> 4, col = lane & 15;
#pragma unroll
  for (int r = 0; r < 8; ++r) {
    float inv = 1.0f / lrow[r];
    int qrow = qb * 16 + r + 8 * hi;
    size_t base = ((size_t)b * SEQ + qrow) * DIM + h * HEAD_DIM + col;
    attn_out[base]      = (__bf16)(o0[r] * inv);
    attn_out[base + 16] = (__bf16)(o1[r] * inv);
    attn_out[base + 32] = (__bf16)(o2[r] * inv);
    attn_out[base + 48] = (__bf16)(o3[r] * inv);
  }
}

// ------------------------------- launcher ----------------------------------

extern "C" void kernel_launch(void* const* d_in, const int* in_sizes, int n_in,
                              void* d_out, int out_size, void* d_ws,
                              size_t ws_size, hipStream_t stream) {
  const float* x      = (const float*)d_in[0];
  const int*   mask   = (const int*)d_in[1];
  const float* w_qkv  = (const float*)d_in[2];
  const float* w_proj = (const float*)d_in[3];
  const float* b_proj = (const float*)d_in[4];
  float* out = (float*)d_out;

  const size_t M = (size_t)BATCH * SEQ;  // 16384

  char* ws = (char*)d_ws;
  size_t off = 0;
  auto carve = [&](size_t bytes) -> char* {
    char* p = ws + off;
    off += (bytes + 255) & ~(size_t)255;
    return p;
  };
  __bf16* xb     = (__bf16*)carve(M * DIM * 2);
  __bf16* wqkvb  = (__bf16*)carve((size_t)DIM * 3 * DIM * 2);
  __bf16* wprojb = (__bf16*)carve((size_t)DIM * DIM * 2);
  __bf16* qw     = (__bf16*)carve(M * DIM * 2);
  __bf16* kw     = (__bf16*)carve(M * DIM * 2);
  __bf16* vw     = (__bf16*)carve(M * DIM * 2);
  __bf16* attn   = (__bf16*)carve(M * DIM * 2);
  (void)ws_size; (void)in_sizes; (void)n_in; (void)out_size;

  // 0) fp32 -> bf16 conversions
  int nx = (int)(M * DIM);
  cvt_f32_bf16_kernel<<<(nx + 255) / 256, 256, 0, stream>>>(x, xb, nx);
  int nq = DIM * 3 * DIM;
  cvt_f32_bf16_kernel<<<(nq + 255) / 256, 256, 0, stream>>>(w_qkv, wqkvb, nq);
  int np = DIM * DIM;
  cvt_f32_bf16_kernel<<<(np + 255) / 256, 256, 0, stream>>>(w_proj, wprojb, np);

  // 1) QKV projection GEMM (16384x768 @ 768x2304) + scatter epilogue
  {
    int nTilesN = (3 * DIM) / 128;                     // 18
    int nTilesM = (int)(M / 128);                      // 128
    gemm_bf16_kernel<<<nTilesM * nTilesN, 256, 0, stream>>>(
        xb, wqkvb, DIM, 3 * DIM, nTilesN, /*mode=*/0, qw, kw, vw, nullptr,
        nullptr);
  }

  // 2) flash attention: one wave per (b, h, 16-query block)
  flash_attn_kernel<<<BATCH * NUM_HEADS * (SEQ / 16), 32, 0, stream>>>(
      qw, kw, vw, mask, attn);

  // 3) output projection GEMM (16384x768 @ 768x768) + bias, fp32 out
  {
    int nTilesN = DIM / 128;                           // 6
    int nTilesM = (int)(M / 128);                      // 128
    gemm_bf16_kernel<<<nTilesM * nTilesN, 256, 0, stream>>>(
        attn, wprojb, DIM, DIM, nTilesN, /*mode=*/1, nullptr, nullptr, nullptr,
        b_proj, out);
  }
}
